// activated_att_59682865545576
// MI455X (gfx1250) — matrix operand order repaired and checked by reading, not compiled, not run on hardware
//
#include <hip/hip_runtime.h>
#include <hip/hip_bf16.h>
#include <math.h>

typedef __attribute__((ext_vector_type(16))) _Float16 v16h;
typedef __attribute__((ext_vector_type(8)))  _Float16 v8h;
typedef __attribute__((ext_vector_type(8)))  float    v8f;

#define HEADS 4
#define HD    32
#define CIN   256
#define NCOL  416   // 128(q) + 128(k) + 128(v) + 32(skip)
#define NT    26    // NCOL / 16

union FragAB { v16h f; v8h h[2]; };

// ---------------------------------------------------------------- utilities
__global__ void zero_f32(float* __restrict__ p, size_t n) {
  size_t i = (size_t)blockIdx.x * blockDim.x + threadIdx.x;
  size_t stride = (size_t)gridDim.x * blockDim.x;
  for (; i < n; i += stride) p[i] = 0.0f;
}

__global__ void cvt_x_f16(const float* __restrict__ x, _Float16* __restrict__ xh, size_t n) {
  size_t i = (size_t)blockIdx.x * blockDim.x + threadIdx.x;
  size_t stride = (size_t)gridDim.x * blockDim.x;
  for (; i < n; i += stride) xh[i] = (_Float16)x[i];
}

// Build W^T (NCOL x CIN, f16) and concatenated bias (NCOL, f32)
__global__ void build_w(const float* __restrict__ Wq, const float* __restrict__ Wk,
                        const float* __restrict__ Wv, const float* __restrict__ Ws,
                        const float* __restrict__ bq, const float* __restrict__ bk,
                        const float* __restrict__ bv, const float* __restrict__ bs,
                        _Float16* __restrict__ WhT, float* __restrict__ bias) {
  int i = blockIdx.x * blockDim.x + threadIdx.x;
  if (i < NCOL * CIN) {
    int c = i / CIN;
    int kk = i - c * CIN;
    float w;
    if (c < 128)      w = Wq[kk * 128 + c];
    else if (c < 256) w = Wk[kk * 128 + (c - 128)];
    else if (c < 384) w = Wv[kk * 128 + (c - 256)];
    else              w = Ws[kk * 32  + (c - 384)];
    WhT[i] = (_Float16)w;
  }
  if (i < NCOL) {
    float b;
    if (i < 128)      b = bq[i];
    else if (i < 256) b = bk[i - 128];
    else if (i < 384) b = bv[i - 256];
    else              b = bs[i - 384];
    bias[i] = b;
  }
}

// ---------------------------------------------------------------- fused projection GEMM
// C[N x 416] = Xh[N x 256] @ WhT^T + bias, one 16x16 tile per wave32,
// v_wmma_f32_16x16x32_f16, K loop of 8.
__global__ void __launch_bounds__(256)
gemm_qkvs(const _Float16* __restrict__ Xh, const _Float16* __restrict__ WhT,
          const float* __restrict__ bias,
          float* __restrict__ qb, float* __restrict__ kb,
          float* __restrict__ vb, float* __restrict__ sk,
          int n, int mTiles) {
  int wave = (int)(((size_t)blockIdx.x * blockDim.x + threadIdx.x) >> 5);
  if (wave >= mTiles * NT) return;           // uniform per wave: EXEC stays all-ones for WMMA
  int mt = wave / NT;
  int nt = wave - mt * NT;
  int lane = threadIdx.x & 31;
  int g  = lane >> 4;                        // lane half (ISA fragment layout)
  int ln = lane & 15;

  int arow = mt * 16 + ln;
  if (arow >= n) arow = n - 1;               // clamp (stores guarded below)
  const _Float16* A = Xh + (size_t)arow * CIN;
  int col = nt * 16 + ln;
  const _Float16* B = WhT + (size_t)col * CIN;

  v8f acc;
  float b0 = bias[col];
#pragma unroll
  for (int r = 0; r < 8; ++r) acc[r] = b0;

#pragma unroll
  for (int k0 = 0; k0 < CIN; k0 += 32) {
    FragAB a, b;
    // 16-bit A 16x32: lanes 0-15 -> K 0..7 / 16..23 ; lanes 16-31 -> K 8..15 / 24..31
    a.h[0] = *(const v8h*)(A + k0 + g * 8);
    a.h[1] = *(const v8h*)(A + k0 + 16 + g * 8);
    // 16-bit B 32x16: lanes 0-15 -> K 0..7 / 16..23 ; lanes 16-31 -> K 8..15 / 24..31 (col = lane&15)
    b.h[0] = *(const v8h*)(B + k0 + g * 8);
    b.h[1] = *(const v8h*)(B + k0 + 16 + g * 8);
    if (k0 + 32 < CIN) {
      __builtin_prefetch(A + k0 + 32, 0, 1);   // global_prefetch_b8
    }
    acc = __builtin_amdgcn_wmma_f32_16x16x32_f16(
        /*neg_a=*/false, a.f, /*neg_b=*/false, b.f,
        /*c_mod=*/(short)0, acc, /*reuse_a=*/false, /*reuse_b=*/false);
  }

  // D layout: lane holds column `col`, VGPR r holds row mt*16 + g*8 + r
#pragma unroll
  for (int r = 0; r < 8; ++r) {
    int row = mt * 16 + g * 8 + r;
    if (row >= n) continue;
    float val = acc[r];
    if (col < 128)      qb[(size_t)row * 128 + col]        = val;
    else if (col < 256) kb[(size_t)row * 128 + (col - 128)] = val;
    else if (col < 384) vb[(size_t)row * 128 + (col - 256)] = val;
    else                sk[(size_t)row * HD  + (col - 384)] = val;
  }
}

// ---------------------------------------------------------------- edge phase
// One wave32 per edge: lane c handles channel c of each head.
// Softmax shift dropped (shift-invariant; logits ~N(0,1) -> no overflow).
__global__ void __launch_bounds__(256)
edge_logits(const int* __restrict__ ei,
            const float* __restrict__ qb, const float* __restrict__ kb,
            float* __restrict__ ee, float* __restrict__ denom, int nE) {
  int wave = (int)(((size_t)blockIdx.x * blockDim.x + threadIdx.x) >> 5);
  if (wave >= nE) return;
  int lane = threadIdx.x & 31;
  int src = ei[wave];
  int dst = ei[nE + wave];
  const float* q  = qb + (size_t)dst * 128;
  const float* kk = kb + (size_t)src * 128;
  float p0 = q[lane]      * kk[lane];
  float p1 = q[32 + lane] * kk[32 + lane];
  float p2 = q[64 + lane] * kk[64 + lane];
  float p3 = q[96 + lane] * kk[96 + lane];
#pragma unroll
  for (int off = 16; off > 0; off >>= 1) {
    p0 += __shfl_xor(p0, off, 32);
    p1 += __shfl_xor(p1, off, 32);
    p2 += __shfl_xor(p2, off, 32);
    p3 += __shfl_xor(p3, off, 32);
  }
  const float scale = 0.17677669529663687f;   // 1/sqrt(32)
  float e0 = __expf(p0 * scale);
  float e1 = __expf(p1 * scale);
  float e2 = __expf(p2 * scale);
  float e3 = __expf(p3 * scale);
  float mine = (lane == 0) ? e0 : (lane == 1) ? e1 : (lane == 2) ? e2 : e3;
  if (lane < 4) {
    ee[(size_t)wave * 4 + lane] = mine;
    atomicAdd(&denom[(size_t)dst * 4 + lane], mine);
  }
}

// Weighted scatter with head-mean fused: out[dst,c] += 0.25 * sum_h alpha_h * v[src,h,c]
__global__ void __launch_bounds__(256)
edge_scatter(const int* __restrict__ ei, const float* __restrict__ vb,
             const float* __restrict__ ee, const float* __restrict__ denom,
             float* __restrict__ accb, int nE) {
  int wave = (int)(((size_t)blockIdx.x * blockDim.x + threadIdx.x) >> 5);
  if (wave >= nE) return;
  int lane = threadIdx.x & 31;
  int src = ei[wave];
  int dst = ei[nE + wave];
  float a = 0.0f;
  if (lane < 4) a = ee[(size_t)wave * 4 + lane] / denom[(size_t)dst * 4 + lane];
  float a0 = __shfl(a, 0, 32);
  float a1 = __shfl(a, 1, 32);
  float a2 = __shfl(a, 2, 32);
  float a3 = __shfl(a, 3, 32);
  const float* v = vb + (size_t)src * 128;
  float s = 0.25f * (a0 * v[lane] + a1 * v[32 + lane] +
                     a2 * v[64 + lane] + a3 * v[96 + lane]);
  atomicAdd(&accb[(size_t)dst * HD + lane], s);
}

// out = leakyrelu(acc + skip, 0.1)
__global__ void finalize_k(const float* __restrict__ accb, const float* __restrict__ sk,
                           float* __restrict__ out, size_t n) {
  size_t i = (size_t)blockIdx.x * blockDim.x + threadIdx.x;
  size_t stride = (size_t)gridDim.x * blockDim.x;
  for (; i < n; i += stride) {
    float v = accb[i] + sk[i];
    out[i] = v > 0.0f ? v : 0.1f * v;
  }
}

// ---------------------------------------------------------------- launch
extern "C" void kernel_launch(void* const* d_in, const int* in_sizes, int n_in,
                              void* d_out, int out_size, void* d_ws, size_t ws_size,
                              hipStream_t stream) {
  const float* x  = (const float*)d_in[0];
  const int*   ei = (const int*)  d_in[1];
  const float* Wq = (const float*)d_in[2];
  const float* bq = (const float*)d_in[3];
  const float* Wk = (const float*)d_in[4];
  const float* bk = (const float*)d_in[5];
  const float* Wv = (const float*)d_in[6];
  const float* bv = (const float*)d_in[7];
  const float* Ws = (const float*)d_in[8];
  const float* bs = (const float*)d_in[9];
  float* out = (float*)d_out;

  int n  = in_sizes[0] / CIN;   // 50000
  int nE = in_sizes[1] / 2;     // 800000

  char* base = (char*)d_ws;
  size_t off = 0;
  auto take = [&](size_t bytes) -> char* {
    char* p = base + off;
    off = (off + bytes + 255) & ~(size_t)255;
    return p;
  };
  _Float16* Xh  = (_Float16*)take((size_t)n * CIN * 2);
  _Float16* WhT = (_Float16*)take((size_t)NCOL * CIN * 2);
  float* bias   = (float*)take((size_t)NCOL * 4);
  float* qb     = (float*)take((size_t)n * 128 * 4);
  float* kb     = (float*)take((size_t)n * 128 * 4);
  float* vb     = (float*)take((size_t)n * 128 * 4);
  float* sk     = (float*)take((size_t)n * HD * 4);
  float* ee     = (float*)take((size_t)nE * 4 * 4);
  float* denom  = (float*)take((size_t)n * 4 * 4);
  float* accb   = (float*)take((size_t)n * HD * 4);
  (void)ws_size; (void)n_in; (void)out_size;

  // accumulators must be re-zeroed every call (graph replay)
  zero_f32<<<1024, 256, 0, stream>>>(denom, (size_t)n * 4);
  zero_f32<<<1024, 256, 0, stream>>>(accb,  (size_t)n * HD);

  cvt_x_f16<<<4096, 256, 0, stream>>>(x, Xh, (size_t)n * CIN);
  build_w<<<(NCOL * CIN + 255) / 256, 256, 0, stream>>>(Wq, Wk, Wv, Ws, bq, bk, bv, bs, WhT, bias);

  int mTiles = (n + 15) / 16;
  int waves  = mTiles * NT;
  gemm_qkvs<<<(waves + 7) / 8, 256, 0, stream>>>(Xh, WhT, bias, qb, kb, vb, sk, n, mTiles);

  edge_logits<<<(nE + 7) / 8, 256, 0, stream>>>(ei, qb, kb, ee, denom, nE);
  edge_scatter<<<(nE + 7) / 8, 256, 0, stream>>>(ei, vb, ee, denom, accb, nE);

  finalize_k<<<2048, 256, 0, stream>>>(accb, sk, out, (size_t)n * HD);
}
